// CustomStridePatchTSTModel_73547019976750
// MI455X (gfx1250) — matrix operand order repaired and checked by reading, not compiled, not run on hardware
//
#include <hip/hip_runtime.h>
#include <hip/hip_bf16.h>

// ---------------- problem constants (match reference) ----------------
#define Bb      8
#define Ll      128
#define Cc      8
#define Hh      256
#define NHh     8
#define DHh     32            // Hh / NHh
#define FFf     1024
#define NLl     2
#define MAXPAD  15
#define LPp     143           // Ll + MAXPAD
#define BL      (Bb * Ll)     // 1024
#define NMAX0   34
#define NMAX1   16
#define E0      136           // LPp - 8 + 1
#define E1      128           // LPp - 16 + 1
#define MAXNP   34            // max Nmax

typedef __attribute__((ext_vector_type(16))) _Float16 v16h;
typedef __attribute__((ext_vector_type(8)))  _Float16 v8h;
typedef __attribute__((ext_vector_type(8)))  float    v8f;

// ---- CDNA5 async global->LDS staging (ASYNCcnt) via inline asm ----
// Builtin signatures demand LangAS-qualified pointer types we cannot spell
// from HIP, so emit the VFLAT async ops directly. LDS operand is the
// wave-relative LDS byte address == low 32 bits of a generic LDS pointer
#if defined(__gfx1250__)
#define USE_ASYNC_LDS 1
#else
#define USE_ASYNC_LDS 0
#endif

#if USE_ASYNC_LDS
__device__ __forceinline__ void async_ld_lds_b128(const void* gaddr, void* lds) {
    asm volatile("global_load_async_to_lds_b128 %0, %1, off"
                 :
                 : "v"((unsigned)(uintptr_t)lds),
                   "v"((unsigned long long)(uintptr_t)gaddr)
                 : "memory");
}
__device__ __forceinline__ void async_ld_lds_b32(const void* gaddr, void* lds) {
    asm volatile("global_load_async_to_lds_b32 %0, %1, off"
                 :
                 : "v"((unsigned)(uintptr_t)lds),
                   "v"((unsigned long long)(uintptr_t)gaddr)
                 : "memory");
}
__device__ __forceinline__ void wait_asynccnt0() {
    asm volatile("s_wait_asynccnt 0" ::: "memory");
}
#endif

__device__ __forceinline__ float gelu_exact(float x) {
    return 0.5f * x * (1.0f + erff(x * 0.70710678118654752f));
}

// ==================================================================
// WMMA GEMM: out[M,N] = epilogue(A[M,K](f16) @ Bt[N,K]^T(f16))
// Bt is the pre-transposed weight (row n holds K contiguous values).
// Block: 256 threads = 8 waves; 64x64 tile; K stepped by 32 via LDS,
// staged with GLOBAL_LOAD_ASYNC_TO_LDS_B128 on gfx1250.
// ==================================================================
template<bool BIAS, bool GELU, bool RES, bool OUT32, bool OUT16>
__launch_bounds__(256)
__global__ void gemm_wmma_kernel(const _Float16* __restrict__ A,
                                 const _Float16* __restrict__ Bt,
                                 const float*    __restrict__ bias,
                                 const float*    __restrict__ res,
                                 float*          __restrict__ out32,
                                 _Float16*       __restrict__ out16,
                                 int M, int N, int K)
{
    __shared__ __align__(16) _Float16 lsa[64 * 32];
    __shared__ __align__(16) _Float16 lsb[64 * 32];

    const int tid  = threadIdx.x;
    const int lane = tid & 31;
    const int wave = tid >> 5;
    const int l15  = lane & 15;
    const int hlf  = lane >> 4;          // 0 or 1 (lane group)
    const int srow = tid >> 2;           // staging row 0..63
    const int skp  = (tid & 3) * 8;      // staging k offset (halves)
    const int m0   = (wave & 3) * 16;    // wave M subtile in block
    const int nb   = (wave >> 2) * 32;   // wave N base (2 tiles of 16)

    const long blockM = (long)blockIdx.x * 64;
    const long blockN = (long)blockIdx.y * 64;

    const _Float16* Arow = A  + (blockM + srow) * (long)K;
    const _Float16* Brow = Bt + (blockN + srow) * (long)K;

    v8f acc0 = {};
    v8f acc1 = {};

    for (int k0 = 0; k0 < K; k0 += 32) {
        __syncthreads();                 // LDS reuse: prior reads done
#if USE_ASYNC_LDS
        async_ld_lds_b128(Arow + k0 + skp, &lsa[srow * 32 + skp]);
        async_ld_lds_b128(Brow + k0 + skp, &lsb[srow * 32 + skp]);
        if (k0 + 32 < K) {               // global_prefetch_b8 next tile
            __builtin_prefetch(Arow + k0 + 32 + skp, 0, 1);
            __builtin_prefetch(Brow + k0 + 32 + skp, 0, 1);
        }
        wait_asynccnt0();
#else
        *reinterpret_cast<v8h*>(&lsa[srow * 32 + skp]) =
            *reinterpret_cast<const v8h*>(Arow + k0 + skp);
        *reinterpret_cast<v8h*>(&lsb[srow * 32 + skp]) =
            *reinterpret_cast<const v8h*>(Brow + k0 + skp);
        if (k0 + 32 < K) {
            __builtin_prefetch(Arow + k0 + 32 + skp, 0, 1);
            __builtin_prefetch(Brow + k0 + 32 + skp, 0, 1);
        }
#endif
        __syncthreads();

        // A fragment (16x32 f16): lanes 0-15 rows m0..m0+15 with K 0-7/16-23,
        // lanes 16-31 same rows with K 8-15/24-31.
        v16h af;
        {
            v8h lo = *reinterpret_cast<const v8h*>(&lsa[(m0 + l15) * 32 + hlf * 8]);
            v8h hi = *reinterpret_cast<const v8h*>(&lsa[(m0 + l15) * 32 + 16 + hlf * 8]);
#pragma unroll
            for (int i = 0; i < 8; ++i) { af[i] = lo[i]; af[i + 8] = hi[i]; }
        }
        // B fragments (32x16 f16): lanes 0-15 hold K 0-7/16-23, lanes 16-31 K 8-15/24-31,
        // column = n-base + (lane&15). Bt rows are K-contiguous -> 2x16B loads.
        v16h bf0, bf1;
        {
            v8h a0 = *reinterpret_cast<const v8h*>(&lsb[(nb + l15) * 32 + hlf * 8]);
            v8h a1 = *reinterpret_cast<const v8h*>(&lsb[(nb + l15) * 32 + 16 + hlf * 8]);
            v8h b0 = *reinterpret_cast<const v8h*>(&lsb[(nb + 16 + l15) * 32 + hlf * 8]);
            v8h b1 = *reinterpret_cast<const v8h*>(&lsb[(nb + 16 + l15) * 32 + 16 + hlf * 8]);
#pragma unroll
            for (int i = 0; i < 8; ++i) {
                bf0[i] = a0[i]; bf0[i + 8] = a1[i];
                bf1[i] = b0[i]; bf1[i + 8] = b1[i];
            }
        }
        acc0 = __builtin_amdgcn_wmma_f32_16x16x32_f16(false, af, false, bf0,
                                                      (short)0, acc0, false, false);
        acc1 = __builtin_amdgcn_wmma_f32_16x16x32_f16(false, af, false, bf1,
                                                      (short)0, acc1, false, false);
    }

    // Epilogue. D layout: VGPR r -> M = r (lanes 0-15) / r+8 (lanes 16-31).
    const long orow0 = blockM + m0 + hlf * 8;
#pragma unroll
    for (int r = 0; r < 8; ++r) {
        const long orow = orow0 + r;
        {
            const long ocol = blockN + nb + l15;
            float v = acc0[r];
            if (BIAS)  v += bias[ocol];
            if (GELU)  v  = gelu_exact(v);
            if (RES)   v += res[orow * (long)N + ocol];
            if (OUT32) out32[orow * (long)N + ocol] = v;
            if (OUT16) out16[orow * (long)N + ocol] = (_Float16)v;
        }
        {
            const long ocol = blockN + nb + 16 + l15;
            float v = acc1[r];
            if (BIAS)  v += bias[ocol];
            if (GELU)  v  = gelu_exact(v);
            if (RES)   v += res[orow * (long)N + ocol];
            if (OUT32) out32[orow * (long)N + ocol] = v;
            if (OUT16) out16[orow * (long)N + ocol] = (_Float16)v;
        }
    }
}

// ==================================================================
// LayerNorm over H=256: one wave32 per row, shfl_xor tree reduction.
// ==================================================================
template<bool OUT32, bool OUT16>
__launch_bounds__(256)
__global__ void layernorm_kernel(const float* __restrict__ in,
                                 const float* __restrict__ gamma,
                                 const float* __restrict__ beta,
                                 float*       __restrict__ out32,
                                 _Float16*    __restrict__ out16)
{
    const int  lane = threadIdx.x & 31;
    const int  wrow = threadIdx.x >> 5;
    const long row  = (long)blockIdx.x * 8 + wrow;
    const float* x  = in + row * Hh;

    float vals[8];
    float s = 0.f, s2 = 0.f;
#pragma unroll
    for (int i = 0; i < 8; ++i) {
        float v = x[lane + 32 * i];
        vals[i] = v; s += v; s2 += v * v;
    }
#pragma unroll
    for (int m = 16; m >= 1; m >>= 1) {
        s  += __shfl_xor(s,  m, 32);
        s2 += __shfl_xor(s2, m, 32);
    }
    const float mu   = s * (1.0f / Hh);
    const float var  = s2 * (1.0f / Hh) - mu * mu;
    const float rstd = rsqrtf(var + 1e-5f);
#pragma unroll
    for (int i = 0; i < 8; ++i) {
        const int e = lane + 32 * i;
        const float y = (vals[i] - mu) * rstd * gamma[e] + beta[e];
        if (OUT32) out32[row * Hh + e] = y;
        if (OUT16) out16[row * Hh + e] = (_Float16)y;
    }
}

// ==================================================================
// Attention: one block per (sequence, head). K/V staged in LDS via
// async b32 loads; online softmax with key mask (k < n(l)); f16 out.
// ==================================================================
__launch_bounds__(64)
__global__ void attention_kernel(const float* __restrict__ qb,
                                 const float* __restrict__ kb,
                                 const float* __restrict__ vb,
                                 _Float16*    __restrict__ o16,
                                 int Nmax, int pl, int st)
{
    __shared__ float ks[MAXNP * DHh];
    __shared__ float vs[MAXNP * DHh];

    const int seq  = blockIdx.x;            // 0..BL-1
    const int head = blockIdx.y;            // 0..NH-1
    const int tid  = threadIdx.x;
    const int l    = seq & (Ll - 1);
    int n = (l + MAXPAD - (pl - 1)) / st + 1;
    if (n > Nmax) n = Nmax;

    const long baseRow = (long)seq * Nmax;
    for (int i = tid; i < Nmax * DHh; i += 64) {
        const int  j   = i / DHh;
        const int  d   = i - j * DHh;
        const long src = (baseRow + j) * Hh + head * DHh + d;
#if USE_ASYNC_LDS
        async_ld_lds_b32(kb + src, &ks[i]);
        async_ld_lds_b32(vb + src, &vs[i]);
#else
        ks[i] = kb[src];
        vs[i] = vb[src];
#endif
    }
#if USE_ASYNC_LDS
    wait_asynccnt0();
#endif
    __syncthreads();

    if (tid < Nmax) {
        float q[DHh];
        const long qsrc = (baseRow + tid) * Hh + head * DHh;
#pragma unroll
        for (int d = 0; d < DHh; ++d) q[d] = qb[qsrc + d];

        float m = -1e30f, ssum = 0.f;
        float o[DHh];
#pragma unroll
        for (int d = 0; d < DHh; ++d) o[d] = 0.f;

        const float rs = 0.17677669529663687f;   // 1/sqrt(DH)
        for (int j = 0; j < Nmax; ++j) {
            float sc;
            if (j < n) {
                sc = 0.f;
#pragma unroll
                for (int d = 0; d < DHh; ++d) sc += q[d] * ks[j * DHh + d];
                sc *= rs;
            } else {
                sc = -1e9f;
            }
            const float nm = fmaxf(m, sc);
            const float f  = __expf(m - nm);
            const float e  = __expf(sc - nm);
            ssum = ssum * f + e;
#pragma unroll
            for (int d = 0; d < DHh; ++d) o[d] = o[d] * f + e * vs[j * DHh + d];
            m = nm;
        }
        const float inv = 1.0f / ssum;
#pragma unroll
        for (int d = 0; d < DHh; ++d)
            o16[(baseRow + tid) * Hh + head * DHh + d] = (_Float16)(o[d] * inv);
    }
}

// ---------------- small prep / gather kernels ----------------

// W[K,N] f32 -> Wt[N,K] f16 (transpose + convert)
__global__ void wcvt_t_kernel(const float* __restrict__ w,
                              _Float16* __restrict__ wt, int K, int N)
{
    const int i = blockIdx.x * blockDim.x + threadIdx.x;
    if (i < K * N) {
        const int k = i / N, n = i - k * N;
        wt[(long)n * K + k] = (_Float16)w[i];
    }
}

// windows16[(b*E+e), p*C+c] = f16(x_padded[b, e+p, c])
__global__ void windows_kernel(const float* __restrict__ x,
                               _Float16* __restrict__ win, int E, int pl)
{
    const int total = Bb * E * pl * Cc;
    const int i = blockIdx.x * blockDim.x + threadIdx.x;
    if (i < total) {
        const int c = i % Cc;
        const int p = (i / Cc) % pl;
        const int e = (i / (Cc * pl)) % E;
        const int b = i / (Cc * pl * E);
        const int t = e + p;
        const float v = (t >= MAXPAD) ? x[((long)b * Ll + (t - MAXPAD)) * Cc + c] : 0.f;
        win[i] = (_Float16)v;
    }
}

// seq[(b*L+l)*Nmax + k, h] = tokens[b, clip(ends-(pl-1)), h] + pos[k, h]
__global__ void seqbuild_kernel(const float* __restrict__ tokens,
                                const float* __restrict__ pos,
                                float* __restrict__ seq,
                                int Nmax, int pl, int st, int E)
{
    const int total = BL * Nmax * Hh;
    const int i = blockIdx.x * blockDim.x + threadIdx.x;
    if (i < total) {
        const int h = i % Hh;
        const int k = (i / Hh) % Nmax;
        const int s = i / (Hh * Nmax);
        const int l = s & (Ll - 1);
        const int b = s / Ll;
        const int n = (l + MAXPAD - (pl - 1)) / st + 1;
        int j = (l + MAXPAD - (n - 1) * st + k * st) - (pl - 1);
        j = j < 0 ? 0 : (j > E - 1 ? E - 1 : j);
        seq[i] = tokens[((long)b * E + j) * Hh + h] + pos[k * Hh + h];
    }
}

// comb[(b*L+l), scale*H + h] = f16(seq[(b*L+l)*Nmax + (n-1), h])
__global__ void emb_kernel(const float* __restrict__ seq,
                           _Float16* __restrict__ comb,
                           int Nmax, int pl, int st, int scale)
{
    const int total = BL * Hh;
    const int i = blockIdx.x * blockDim.x + threadIdx.x;
    if (i < total) {
        const int h = i % Hh;
        const int s = i / Hh;
        const int l = s & (Ll - 1);
        const int n = (l + MAXPAD - (pl - 1)) / st + 1;
        comb[(long)s * (2 * Hh) + scale * Hh + h] =
            (_Float16)seq[((long)s * Nmax + (n - 1)) * Hh + h];
    }
}

// ==================================================================
// Host-side orchestration
// ==================================================================
static inline int cdiv(int a, int b) { return (a + b - 1) / b; }

extern "C" void kernel_launch(void* const* d_in, const int* in_sizes, int n_in,
                              void* d_out, int out_size, void* d_ws, size_t ws_size,
                              hipStream_t stream)
{
    (void)in_sizes; (void)n_in; (void)out_size; (void)ws_size;

    const float* x       = (const float*)d_in[0];
    const float* wprojF[2] = { (const float*)d_in[1], (const float*)d_in[3] };
    const float* bproj[2]  = { (const float*)d_in[2], (const float*)d_in[4] };
    const float* pos[2]    = { (const float*)d_in[5], (const float*)d_in[6] };
    const float* ln1g = (const float*)d_in[7];
    const float* ln1b = (const float*)d_in[8];
    const float* wqF  = (const float*)d_in[9];   const float* bqF = (const float*)d_in[10];
    const float* wkF  = (const float*)d_in[11];  const float* bkF = (const float*)d_in[12];
    const float* wvF  = (const float*)d_in[13];  const float* bvF = (const float*)d_in[14];
    const float* woF  = (const float*)d_in[15];  const float* boF = (const float*)d_in[16];
    const float* ln2g = (const float*)d_in[17];
    const float* ln2b = (const float*)d_in[18];
    const float* wf1F = (const float*)d_in[19];  const float* bf1 = (const float*)d_in[20];
    const float* wf2F = (const float*)d_in[21];  const float* bf2 = (const float*)d_in[22];
    const float* lnG  = (const float*)d_in[23];
    const float* lnB  = (const float*)d_in[24];
    const float* wu1F = (const float*)d_in[25];  const float* bu1 = (const float*)d_in[26];
    const float* wu2F = (const float*)d_in[27];  const float* bu2 = (const float*)d_in[28];

    // ---- bump allocator over workspace ----
    char*  base = (char*)d_ws;
    size_t pos_  = 0;
    auto alloc = [&](size_t bytes) -> void* {
        void* p = base + pos_;
        pos_ = (pos_ + bytes + 255) & ~(size_t)255;
        return p;
    };
    auto a16 = [&](size_t n) { return (_Float16*)alloc(n * sizeof(_Float16)); };
    auto a32 = [&](size_t n) { return (float*)alloc(n * sizeof(float)); };

    _Float16* wprojT[2] = { a16((size_t)Hh * 64), a16((size_t)Hh * 128) };
    _Float16* wqT   = a16((size_t)NLl * Hh * Hh);
    _Float16* wkT   = a16((size_t)NLl * Hh * Hh);
    _Float16* wvT   = a16((size_t)NLl * Hh * Hh);
    _Float16* woT   = a16((size_t)NLl * Hh * Hh);
    _Float16* wf1T  = a16((size_t)NLl * FFf * Hh);   // [FF][H] per layer
    _Float16* wf2T  = a16((size_t)NLl * Hh * FFf);   // [H][FF] per layer
    _Float16* wu1T  = a16((size_t)Hh * 2 * Hh);      // [H][2H]
    _Float16* wu2T  = a16((size_t)Hh * Hh);

    const size_t M0max = (size_t)BL * NMAX0;         // 34816 rows
    _Float16* win16 = a16((size_t)Bb * E1 * 16 * Cc);
    float*    tokens = a32((size_t)Bb * E0 * Hh);
    float*    seq   = a32(M0max * Hh);
    _Float16* s2h   = a16(M0max * Hh);
    float*    qbuf  = a32(M0max * Hh);
    float*    kbuf  = a32(M0max * Hh);
    float*    vbuf  = a32(M0max * Hh);
    _Float16* o16   = a16(M0max * Hh);
    _Float16* ffh   = a16(M0max * FFf);
    _Float16* comb  = a16((size_t)BL * 2 * Hh);
    _Float16* fu1h  = a16((size_t)BL * Hh);
    float*    fu2   = a32((size_t)BL * Hh);

    const dim3 blk(256);
    auto tlaunch = [&](const float* w, _Float16* wt, int K, int N) {
        wcvt_t_kernel<<<cdiv(K * N, 256), blk, 0, stream>>>(w, wt, K, N);
    };

    // ---- one-time (per call) weight transpose+convert to f16 ----
    tlaunch(wprojF[0], wprojT[0], 8 * Cc, Hh);
    tlaunch(wprojF[1], wprojT[1], 16 * Cc, Hh);
    for (int li = 0; li < NLl; ++li) {
        tlaunch(wqF + (size_t)li * Hh * Hh,  wqT + (size_t)li * Hh * Hh,  Hh, Hh);
        tlaunch(wkF + (size_t)li * Hh * Hh,  wkT + (size_t)li * Hh * Hh,  Hh, Hh);
        tlaunch(wvF + (size_t)li * Hh * Hh,  wvT + (size_t)li * Hh * Hh,  Hh, Hh);
        tlaunch(woF + (size_t)li * Hh * Hh,  woT + (size_t)li * Hh * Hh,  Hh, Hh);
        tlaunch(wf1F + (size_t)li * Hh * FFf, wf1T + (size_t)li * FFf * Hh, Hh, FFf);
        tlaunch(wf2F + (size_t)li * FFf * Hh, wf2T + (size_t)li * Hh * FFf, FFf, Hh);
    }
    tlaunch(wu1F, wu1T, 2 * Hh, Hh);
    tlaunch(wu2F, wu2T, Hh, Hh);

    // ---- per-scale pipeline ----
    const int PL[2] = { 8, 16 };
    const int ST[2] = { 4, 8 };
    const int NM[2] = { NMAX0, NMAX1 };
    const int EE[2] = { E0, E1 };

    for (int sc = 0; sc < 2; ++sc) {
        const int pl = PL[sc], st = ST[sc], Nmax = NM[sc], E = EE[sc];
        const int Kp = pl * Cc;                 // 64 / 128
        const int Mtok = Bb * E;                // 1088 / 1024 (both %64==0)
        const int M0   = BL * Nmax;             // 34816 / 16384

        // patch windows -> tokens = windows @ w_proj + b_proj
        windows_kernel<<<cdiv(Bb * E * pl * Cc, 256), blk, 0, stream>>>(x, win16, E, pl);
        gemm_wmma_kernel<true, false, false, true, false>
            <<<dim3(Mtok / 64, Hh / 64), blk, 0, stream>>>(
                win16, wprojT[sc], bproj[sc], nullptr, tokens, nullptr, Mtok, Hh, Kp);

        // seq = gather(tokens) + pos
        seqbuild_kernel<<<cdiv(M0 * Hh, 256), blk, 0, stream>>>(
            tokens, pos[sc], seq, Nmax, pl, st, E);

        for (int li = 0; li < NLl; ++li) {
            const _Float16* wqTl = wqT + (size_t)li * Hh * Hh;
            const _Float16* wkTl = wkT + (size_t)li * Hh * Hh;
            const _Float16* wvTl = wvT + (size_t)li * Hh * Hh;
            const _Float16* woTl = woT + (size_t)li * Hh * Hh;

            // s2 = LN1(seq) -> f16
            layernorm_kernel<false, true><<<M0 / 8, blk, 0, stream>>>(
                seq, ln1g + li * Hh, ln1b + li * Hh, nullptr, s2h);

            // Q, K, V
            gemm_wmma_kernel<true, false, false, true, false>
                <<<dim3(M0 / 64, Hh / 64), blk, 0, stream>>>(
                    s2h, wqTl, bqF + li * Hh, nullptr, qbuf, nullptr, M0, Hh, Hh);
            gemm_wmma_kernel<true, false, false, true, false>
                <<<dim3(M0 / 64, Hh / 64), blk, 0, stream>>>(
                    s2h, wkTl, bkF + li * Hh, nullptr, kbuf, nullptr, M0, Hh, Hh);
            gemm_wmma_kernel<true, false, false, true, false>
                <<<dim3(M0 / 64, Hh / 64), blk, 0, stream>>>(
                    s2h, wvTl, bvF + li * Hh, nullptr, vbuf, nullptr, M0, Hh, Hh);

            // masked attention -> o16 (f16)
            attention_kernel<<<dim3(BL, NHh), dim3(64), 0, stream>>>(
                qbuf, kbuf, vbuf, o16, Nmax, pl, st);

            // seq += o @ wo + bo
            gemm_wmma_kernel<true, false, true, true, false>
                <<<dim3(M0 / 64, Hh / 64), blk, 0, stream>>>(
                    o16, woTl, boF + li * Hh, seq, seq, nullptr, M0, Hh, Hh);

            // s2 = LN2(seq) -> f16
            layernorm_kernel<false, true><<<M0 / 8, blk, 0, stream>>>(
                seq, ln2g + li * Hh, ln2b + li * Hh, nullptr, s2h);

            // ffh = gelu(s2 @ w_ff1 + b_ff1) -> f16
            gemm_wmma_kernel<true, true, false, false, true>
                <<<dim3(M0 / 64, FFf / 64), blk, 0, stream>>>(
                    s2h, wf1T + (size_t)li * FFf * Hh, bf1 + li * FFf,
                    nullptr, nullptr, ffh, M0, FFf, Hh);

            // seq += ffh @ w_ff2 + b_ff2
            gemm_wmma_kernel<true, false, true, true, false>
                <<<dim3(M0 / 64, Hh / 64), blk, 0, stream>>>(
                    ffh, wf2T + (size_t)li * Hh * FFf, bf2 + li * Hh,
                    seq, seq, nullptr, M0, Hh, FFf);
        }

        // last real patch embedding -> comb[:, sc*H : (sc+1)*H]
        emb_kernel<<<cdiv(BL * Hh, 256), blk, 0, stream>>>(seq, comb, Nmax, pl, st, sc);
    }

    // fused = gelu(comb @ w_fus1 + b_fus1) -> f16
    gemm_wmma_kernel<true, true, false, false, true>
        <<<dim3(BL / 64, Hh / 64), blk, 0, stream>>>(
            comb, wu1T, bu1, nullptr, nullptr, fu1h, BL, Hh, 2 * Hh);
    // fu2 = fused @ w_fus2 + b_fus2 -> f32
    gemm_wmma_kernel<true, false, false, true, false>
        <<<dim3(BL / 64, Hh / 64), blk, 0, stream>>>(
            fu1h, wu2T, bu2, nullptr, fu2, nullptr, BL, Hh, Hh);
    // final LN -> d_out (f32)
    layernorm_kernel<true, false><<<BL / 8, blk, 0, stream>>>(
        fu2, lnG, lnB, (float*)d_out, nullptr);
}
